// ModalityUntiedAttention_1477468749958
// MI455X (gfx1250) — compile-verified, matrix operand order checked by reading, not run on hardware
//
#include <hip/hip_runtime.h>

#define E_    2
#define HQ_   16
#define HK_   8
#define HD_   128
#define DIM_  2048
#define BS_   2
#define SEQ_  2048
#define NTOK  (BS_ * SEQ_)          // 4096 tokens
#define QKVC  (HQ_*HD_ + 2*HK_*HD_) // 4096 fused qkv columns
#define EPS_  1e-6f

typedef __attribute__((ext_vector_type(16))) __bf16 v16bf;
typedef __attribute__((ext_vector_type(8)))  __bf16 v8bf;
typedef __attribute__((ext_vector_type(8)))  float  v8f;

// ---- CDNA5 async global->LDS copy (ASYNCcnt path), per CDNA5 ISA ----
// VDST operand carries the LDS byte offset; for shared-aperture flat pointers
// the low 32 bits are exactly that offset.
__device__ __forceinline__ void async_copy16(const void* g, void* l) {
    unsigned lofs = (unsigned)(size_t)l;
    asm volatile("global_load_async_to_lds_b128 %0, %1, off"
                 :: "v"(lofs), "v"(g) : "memory");
}
#define ASYNC_WAIT_0() asm volatile("s_wait_asynccnt 0x0" ::: "memory")
#define ASYNC_WAIT_4() asm volatile("s_wait_asynccnt 0x4" ::: "memory")
#define ASYNC_WAIT_8() asm volatile("s_wait_asynccnt 0x8" ::: "memory")

// Load a 16x32 bf16 WMMA fragment (A-layout; B-fragments load identically from
// the transposed operand). Per ISA 16-bit A layout: lanes 0-15 hold row M=lane,
// K chunks {0..7,16..23}; lanes 16-31 hold row M=lane-16, K chunks {8..15,24..31}.
__device__ __forceinline__ v16bf load_frag(const __bf16* base, int rstride, int lane) {
    const int r  = lane & 15;
    const int hs = (lane >> 4) << 3;   // 0 or 8
    const __bf16* p = base + (size_t)r * rstride + hs;
    union { v16bf v; v8bf h[2]; } u;
    __builtin_memcpy(&u.h[0], p, 16);
    __builtin_memcpy(&u.h[1], p + 16, 16);
    return u.v;
}

__device__ __forceinline__ v8f wmma_bf16(v16bf a, v16bf b, v8f c) {
    // (neg_a, A, neg_b, B, c_mod, C, reuse_a, reuse_b)
    return __builtin_amdgcn_wmma_f32_16x16x32_bf16(false, a, false, b, (short)0, c, false, false);
}

// ---------------- prep kernels ----------------

__global__ void cvt_x_kernel(const float* __restrict__ src, __bf16* __restrict__ dst) {
    size_t i = (size_t)blockIdx.x * blockDim.x + threadIdx.x;
    dst[i] = (__bf16)src[i];
}

// dst layout: [e][c:4096][k:2048]  (transposed, fused wq|wk|wv)
__global__ void build_wqkv_kernel(const float* __restrict__ wq, const float* __restrict__ wk,
                                  const float* __restrict__ wv, __bf16* __restrict__ dst) {
    size_t i = (size_t)blockIdx.x * blockDim.x + threadIdx.x;
    int k = (int)(i & (DIM_ - 1));
    size_t t = i >> 11;
    int c = (int)(t & (QKVC - 1));
    int e = (int)(t >> 12);
    float v;
    if (c < HQ_*HD_)                 v = wq[((size_t)e * DIM_ + k) * (HQ_*HD_) + c];
    else if (c < HQ_*HD_ + HK_*HD_)  v = wk[((size_t)e * DIM_ + k) * (HK_*HD_) + (c - HQ_*HD_)];
    else                             v = wv[((size_t)e * DIM_ + k) * (HK_*HD_) + (c - HQ_*HD_ - HK_*HD_)];
    dst[i] = (__bf16)v;
}

// dst layout: [e][c:2048][k:2048]  (transposed wo)
__global__ void build_wo_kernel(const float* __restrict__ wo, __bf16* __restrict__ dst) {
    size_t i = (size_t)blockIdx.x * blockDim.x + threadIdx.x;
    int kk = (int)(i & (DIM_ - 1));
    size_t t = i >> 11;
    int c = (int)(t & (DIM_ - 1));
    int e = (int)(t >> 11);
    dst[i] = (__bf16)wo[((size_t)e * (HQ_*HD_) + kk) * DIM_ + c];
}

// ---------------- expert-gated GEMM:  C[n, :] = A[n, :] @ Wt[mod[n]]^T ----------------
// A: [NTOK][K] bf16. Wt: [E][ncols][K] bf16. C: [NTOK][ncols] f32.
// Block = 128 thr (4 waves), tile 64 rows x 64 cols. B panel double-buffered in LDS
// via async global->LDS DMA; A fragments loaded direct (wave-private rows).
__global__ __launch_bounds__(128) void gemm_moe_kernel(
    const __bf16* __restrict__ A, const __bf16* __restrict__ Wt,
    const int* __restrict__ mods, float* __restrict__ C, int K, int ncols) {
    const int tid  = threadIdx.x;
    const int lane = tid & 31;
    const int wave = tid >> 5;
    const int col0 = blockIdx.x * 64;
    const int row0 = blockIdx.y * 64 + wave * 16;
    const int myMod = mods[row0 + (lane & 15)];   // each lane owns one A row
    v8f acc[4] = {};
    const __bf16* Abase = A + (size_t)row0 * K;
    const v16bf zf = {};

    // [buf][expert][col(64)][k(32)] : 8KB per buffer
    __shared__ __align__(16) __bf16 Bs[2][E_][64][32];

    // One batch = 512 x 16B chunks = 4 issues per thread.
    auto issueB = [&](int buf, int k0) {
        #pragma unroll
        for (int j = 0; j < 4; ++j) {
            int ci = j * 128 + tid;
            int kc = ci & 3, rowi = ci >> 2;
            int e = rowi >> 6, c = rowi & 63;
            async_copy16(Wt + (size_t)(e * ncols + col0 + c) * K + k0 + kc * 8,
                         &Bs[buf][e][c][kc * 8]);
        }
    };

    const int nsteps = K / 32;
    issueB(0, 0);
    for (int i = 0; i < nsteps; ++i) {
        const int k0 = i * 32;
        if (i + 1 < nsteps) {
            issueB((i + 1) & 1, k0 + 32);
            __builtin_prefetch(Abase + (size_t)(lane & 15) * K + k0 + 32, 0, 0);
            ASYNC_WAIT_4();           // older batch (4 chunks) complete; newer may fly
        } else {
            ASYNC_WAIT_0();
        }
        __syncthreads();
        v16bf af = load_frag(Abase + k0, K, lane);
        #pragma unroll
        for (int e = 0; e < E_; ++e) {
            v16bf a = af;
            if (myMod != e) a = zf;  // per-row expert gating; sum over e == gather
            #pragma unroll
            for (int c = 0; c < 4; ++c) {
                v16bf bfrag = load_frag(&Bs[i & 1][e][c * 16][0], 32, lane);
                acc[c] = wmma_bf16(a, bfrag, acc[c]);
            }
        }
        __syncthreads();             // protect buffer before next-next issue
    }
    const int moff = (lane >> 4) << 3;
    const int ncol = lane & 15;
    #pragma unroll
    for (int c = 0; c < 4; ++c)
        #pragma unroll
        for (int j = 0; j < 8; ++j)
            C[(size_t)(row0 + j + moff) * ncols + col0 + c * 16 + ncol] = acc[c][j];
}

// ---------------- per-head RMSNorm + RoPE, V transpose ----------------
// grid = NTOK * 32 blocks of 128: hh<16 -> q head, hh<24 -> k head, else v head.
__global__ __launch_bounds__(128) void norm_rope_kernel(
    const float* __restrict__ qkvf, const float* __restrict__ cosp, const float* __restrict__ sinp,
    const float* __restrict__ qnw, const float* __restrict__ knw, const int* __restrict__ mods,
    __bf16* __restrict__ qb, __bf16* __restrict__ kb, __bf16* __restrict__ vbT) {
    const int n  = blockIdx.x >> 5;
    const int hh = blockIdx.x & 31;
    const int t  = threadIdx.x;
    const int s  = n & (SEQ_ - 1);
    const int bi = n >> 11;           // n / SEQ
    const int e  = mods[n];
    __shared__ float red[4];
    if (hh < HQ_ + HK_) {
        const bool isq = hh < HQ_;
        const int head = isq ? hh : hh - HQ_;
        const int coff = isq ? head * HD_ : HQ_*HD_ + head * HD_;
        float v = qkvf[(size_t)n * QKVC + coff + t];
        float ss = v * v;
        #pragma unroll
        for (int off = 16; off > 0; off >>= 1) ss += __shfl_xor(ss, off);
        if ((t & 31) == 0) red[t >> 5] = ss;
        __syncthreads();
        float tot = red[0] + red[1] + red[2] + red[3];
        const float* nw = isq ? qnw : knw;
        float vn = v * rsqrtf(tot * (1.0f / HD_) + EPS_) * nw[e * HD_ + t];
        // RoPE over pairs (2i, 2i+1): partner lives in lane t^1
        float other = __shfl_xor(vn, 1);
        int i = t >> 1;
        float c  = cosp[s * (HD_/2) + i];
        float sn = sinp[s * (HD_/2) + i];
        float t0 = (t & 1) ? other : vn;
        float t1 = (t & 1) ? vn : other;
        float outv = (t & 1) ? (t0 * sn + t1 * c) : (t0 * c - t1 * sn);
        if (isq) qb[(size_t)n * (HQ_*HD_) + head * HD_ + t] = (__bf16)outv;
        else     kb[(size_t)n * (HK_*HD_) + head * HD_ + t] = (__bf16)outv;
    } else {
        const int head = hh - (HQ_ + HK_);
        float v = qkvf[(size_t)n * QKVC + (HQ_+HK_)*HD_ + head * HD_ + t];
        // vbT: [b][hk][d][seq]  (transposed so PV B-fragments load contiguously)
        vbT[((size_t)(bi * HK_ + head) * HD_ + t) * SEQ_ + s] = (__bf16)v;
    }
}

// ---------------- causal flash attention (GQA, n_rep = 2) ----------------
// grid = (SEQ/64, HQ, BS), block 128 (4 waves). Wave strip = 16 q rows.
// K/V tiles double-buffered in LDS via async DMA; all waves run the block-level
// k-loop (barrier-safe); compute is skipped wave-uniformly past the causal bound.
__global__ __launch_bounds__(128) void attn_fwd_kernel(
    const __bf16* __restrict__ qb, const __bf16* __restrict__ kb,
    const __bf16* __restrict__ vbT, __bf16* __restrict__ ob) {
    const int tid  = threadIdx.x;
    const int lane = tid & 31;
    const int wave = tid >> 5;
    const int qt = blockIdx.x, h = blockIdx.y, b = blockIdx.z;
    const int kvh = h >> 1;
    const int q0 = qt * 64 + wave * 16;

    __shared__ __align__(16) __bf16 Ks[2][32][128];   // k-rows x d
    __shared__ __align__(16) __bf16 Vs[2][128][32];   // d x k-rows (from vbT)
    __shared__ __align__(16) __bf16 pS[4][16][32];    // wave-private P relayout

    const __bf16* qbase = qb + ((size_t)(b * SEQ_ + q0) * HQ_ + h) * HD_;
    v16bf qf[4];
    #pragma unroll
    for (int kk = 0; kk < 4; ++kk) qf[kk] = load_frag(qbase + kk * 32, HQ_ * HD_, lane);
    v8f o[8] = {};
    float mrow[8], lrow[8];
    #pragma unroll
    for (int j = 0; j < 8; ++j) { mrow[j] = -1e30f; lrow[j] = 0.f; }
    const int moff = (lane >> 4) << 3;
    const int ncol = lane & 15;
    const float scale = 0.08838834764831845f;  // 1/sqrt(128)

    auto issueKV = [&](int buf, int k0) {
        #pragma unroll
        for (int j = 0; j < 4; ++j) {          // K tile: 512 chunks
            int ci = j * 128 + tid;
            int kr = ci >> 4, ch = ci & 15;
            async_copy16(kb + ((size_t)(b * SEQ_ + k0 + kr) * HK_ + kvh) * HD_ + ch * 8,
                         &Ks[buf][kr][ch * 8]);
        }
        #pragma unroll
        for (int j = 0; j < 4; ++j) {          // V tile: 512 chunks
            int ci = j * 128 + tid;
            int dr = ci >> 2, ch = ci & 3;
            async_copy16(vbT + ((size_t)(b * HK_ + kvh) * HD_ + dr) * SEQ_ + k0 + ch * 8,
                         &Vs[buf][dr][ch * 8]);
        }
    };

    const int nsteps = ((qt * 64 + 63) >> 5) + 1;  // block-level causal bound
    const int mykt   = (q0 + 15) >> 5;             // this wave's causal bound
    issueKV(0, 0);
    for (int kt = 0; kt < nsteps; ++kt) {
        const int k0 = kt * 32;
        if (kt + 1 < nsteps) { issueKV((kt + 1) & 1, k0 + 32); ASYNC_WAIT_8(); }
        else                 { ASYNC_WAIT_0(); }
        __syncthreads();
        if (kt <= mykt) {   // wave-uniform: EXEC all-ones inside
            // S strip = Q (16x128) @ K^T tile (128x32) -> two 16x16 accumulators
            v8f c0 = {}, c1 = {};
            #pragma unroll
            for (int kk = 0; kk < 4; ++kk) {
                v16bf b0 = load_frag(&Ks[kt & 1][0][kk * 32], 128, lane);
                v16bf b1 = load_frag(&Ks[kt & 1][16][kk * 32], 128, lane);
                c0 = wmma_bf16(qf[kk], b0, c0);
                c1 = wmma_bf16(qf[kk], b1, c1);
            }
            // online softmax per row (rows live in 16-lane halves of the C layout)
            #pragma unroll
            for (int j = 0; j < 8; ++j) {
                const int sq = q0 + j + moff;
                float s0 = c0[j] * scale;
                float s1 = c1[j] * scale;
                if (k0 + ncol      > sq) s0 = -1e30f;
                if (k0 + 16 + ncol > sq) s1 = -1e30f;
                float mx = fmaxf(s0, s1);
                #pragma unroll
                for (int off = 8; off > 0; off >>= 1) mx = fmaxf(mx, __shfl_xor(mx, off));
                const float mnew = fmaxf(mrow[j], mx);
                const float sc = __expf(mrow[j] - mnew);
                const float p0 = __expf(s0 - mnew);
                const float p1 = __expf(s1 - mnew);
                float rs = p0 + p1;
                #pragma unroll
                for (int off = 8; off > 0; off >>= 1) rs += __shfl_xor(rs, off);
                lrow[j] = lrow[j] * sc + rs;
                mrow[j] = mnew;
                #pragma unroll
                for (int d = 0; d < 8; ++d) o[d][j] *= sc;
                pS[wave][j + moff][ncol]      = (__bf16)p0;
                pS[wave][j + moff][ncol + 16] = (__bf16)p1;
            }
            // O += P (16x32) @ V tile (32x128); wave-private LDS, no block barrier
            v16bf pf = load_frag(&pS[wave][0][0], 32, lane);
            #pragma unroll
            for (int d = 0; d < 8; ++d) {
                v16bf vf = load_frag(&Vs[kt & 1][d * 16][0], 32, lane);
                o[d] = wmma_bf16(pf, vf, o[d]);
            }
        }
        __syncthreads();
    }
    #pragma unroll
    for (int d = 0; d < 8; ++d)
        #pragma unroll
        for (int j = 0; j < 8; ++j) {
            float val = o[d][j] / lrow[j];
            ob[((size_t)(b * SEQ_ + q0 + j + moff) * HQ_ + h) * HD_ + d * 16 + ncol] = (__bf16)val;
        }
}

// ---------------- final RMSNorm with expert-gated weight ----------------
__global__ __launch_bounds__(256) void out_norm_kernel(
    const float* __restrict__ opre, const float* __restrict__ anw,
    const int* __restrict__ mods, float* __restrict__ out) {
    const int n = blockIdx.x;
    const int t = threadIdx.x;
    const int e = mods[n];
    const float* row = opre + (size_t)n * DIM_;
    float v[8];
    float ss = 0.f;
    #pragma unroll
    for (int i = 0; i < 8; ++i) { v[i] = row[t + i * 256]; ss += v[i] * v[i]; }
    #pragma unroll
    for (int off = 16; off > 0; off >>= 1) ss += __shfl_xor(ss, off);
    __shared__ float red[8];
    if ((t & 31) == 0) red[t >> 5] = ss;
    __syncthreads();
    float tot = 0.f;
    #pragma unroll
    for (int i = 0; i < 8; ++i) tot += red[i];
    const float inv = rsqrtf(tot * (1.0f / DIM_) + EPS_);
    float* orow = out + (size_t)n * DIM_;
    #pragma unroll
    for (int i = 0; i < 8; ++i)
        orow[t + i * 256] = v[i] * inv * anw[e * DIM_ + t + i * 256];
}

// ---------------- launch ----------------

extern "C" void kernel_launch(void* const* d_in, const int* in_sizes, int n_in,
                              void* d_out, int out_size, void* d_ws, size_t ws_size,
                              hipStream_t stream) {
    (void)in_sizes; (void)n_in; (void)out_size; (void)ws_size;
    const float* x   = (const float*)d_in[0];
    const float* fc  = (const float*)d_in[1];
    const float* fs  = (const float*)d_in[2];
    const float* wq  = (const float*)d_in[3];
    const float* wk  = (const float*)d_in[4];
    const float* wv  = (const float*)d_in[5];
    const float* wo  = (const float*)d_in[6];
    const float* qnw = (const float*)d_in[7];
    const float* knw = (const float*)d_in[8];
    const float* anw = (const float*)d_in[9];
    const int*  mods = (const int*)d_in[10];
    float* out = (float*)d_out;

    char* ws = (char*)d_ws;
    size_t off = 0;
    auto take = [&](size_t bytes) -> char* {
        char* p = ws + off;
        off = (off + bytes + 255) & ~(size_t)255;
        return p;
    };
    __bf16* xb   = (__bf16*)take((size_t)NTOK * DIM_ * 2);         // bf16 x
    __bf16* wqkv = (__bf16*)take((size_t)E_ * QKVC * DIM_ * 2);    // [e][c][k] fused qkv W^T
    __bf16* wot  = (__bf16*)take((size_t)E_ * DIM_ * DIM_ * 2);    // [e][c][k] wo^T
    float*  qkvf = (float*) take((size_t)NTOK * QKVC * 4);         // fused qkv, f32
    __bf16* qb   = (__bf16*)take((size_t)NTOK * HQ_ * HD_ * 2);
    __bf16* kb   = (__bf16*)take((size_t)NTOK * HK_ * HD_ * 2);
    __bf16* vbT  = (__bf16*)take((size_t)BS_ * HK_ * HD_ * SEQ_ * 2);
    __bf16* ob   = (__bf16*)take((size_t)NTOK * HQ_ * HD_ * 2);
    float*  opre = (float*) take((size_t)NTOK * DIM_ * 4);

    cvt_x_kernel<<<(NTOK * DIM_) / 256, 256, 0, stream>>>(x, xb);
    build_wqkv_kernel<<<(E_ * QKVC * DIM_) / 256, 256, 0, stream>>>(wq, wk, wv, wqkv);
    build_wo_kernel<<<(E_ * DIM_ * DIM_) / 256, 256, 0, stream>>>(wo, wot);

    gemm_moe_kernel<<<dim3(QKVC / 64, NTOK / 64), 128, 0, stream>>>(
        xb, wqkv, mods, qkvf, DIM_, QKVC);

    norm_rope_kernel<<<NTOK * 32, 128, 0, stream>>>(
        qkvf, fc, fs, qnw, knw, mods, qb, kb, vbT);

    attn_fwd_kernel<<<dim3(SEQ_ / 64, HQ_, BS_), 128, 0, stream>>>(qb, kb, vbT, ob);

    gemm_moe_kernel<<<dim3(DIM_ / 64, NTOK / 64), 128, 0, stream>>>(
        ob, wot, mods, opre, HQ_ * HD_, DIM_);

    out_norm_kernel<<<NTOK, 256, 0, stream>>>(opre, anw, mods, out);
}
